// CausalMoe_33002528702995
// MI455X (gfx1250) — compile-verified, matrix operand order checked by reading, not run on hardware
//
#include <hip/hip_runtime.h>
#include <hip/hip_bf16.h>
#include <cmath>

// ---------------- problem constants ----------------
#define BB 4
#define SS 2048
#define DD 1024
#define FF 4096
#define EE 4
#define TT (BB * SS)           // 8192 tokens

typedef __bf16 v16bf __attribute__((ext_vector_type(16)));
typedef float  v8f   __attribute__((ext_vector_type(8)));
typedef unsigned u32x4 __attribute__((ext_vector_type(4)));
typedef float  f32x2 __attribute__((ext_vector_type(2)));

// explicit global address space: force global_load_b128 (LOADcnt only),
// never generic FLAT (which would also tie up DScnt per ISA 10.1.1)
#define GAS __attribute__((address_space(1)))
typedef const GAS u32x4* gq_t;
typedef const GAS f32x2* gf2_t;

union Frag32 { v16bf v; u32x4 q[2]; };

__device__ __forceinline__ unsigned bf_rne(float f) {
    unsigned u = __float_as_uint(f);
    return (u + 0x7FFFu + ((u >> 16) & 1u)) >> 16;   // round-to-nearest-even bf16
}
__device__ __forceinline__ unsigned pack_bf2(float lo, float hi) {
    return bf_rne(lo) | (bf_rne(hi) << 16);
}

// ---------------------------------------------------------------------------
// Kernel A: fp32 [K][N] weight -> bf16 WMMA B-fragments in ws.
// Fragment (e, ft, kt): 1024 bytes = [lane 0..31][16 halves], lane holds column
// n = ft*16 + (lane&15); half j -> k = kt*32 + j + 8*(j>=8) + 8*(lane>=16).
// ---------------------------------------------------------------------------
__global__ void k_swizzle_w(const float* __restrict__ src, unsigned* __restrict__ dst,
                            int K, int N) {
    int tid  = blockIdx.x * blockDim.x + threadIdx.x;
    int lane = tid & 31;
    int frag = tid >> 5;                 // e-major: ((e*NT + ft)*KT + kt)
    int KT = K >> 5, NT = N >> 4;
    int kt = frag % KT;
    int tmp = frag / KT;
    int ft = tmp % NT;
    int e  = tmp / NT;
    int n  = ft * 16 + (lane & 15);
    int hi = (lane >> 4) * 8;
    const float* s = src + (size_t)e * K * N;
    unsigned w[8];
#pragma unroll
    for (int i = 0; i < 8; ++i) {
        int k0 = kt * 32 + 2 * i + ((i >= 4) ? 8 : 0) + hi;
        float a = s[(size_t)k0 * N + n];
        float b = s[(size_t)(k0 + 1) * N + n];
        w[i] = pack_bf2(a, b);
    }
    u32x4* d = (u32x4*)(dst + (size_t)frag * 256 + lane * 8);
    d[0] = (u32x4){w[0], w[1], w[2], w[3]};
    d[1] = (u32x4){w[4], w[5], w[6], w[7]};
}

// ---------------------------------------------------------------------------
// Kernel B: router raw logits r[tok][e] = x[tok]·router_w[:,e]; wave per token.
// ---------------------------------------------------------------------------
__global__ void k_router(const float* __restrict__ x, const float* __restrict__ rw,
                         float4* __restrict__ r) {
    int tok  = blockIdx.x * 8 + (threadIdx.x >> 5);
    int lane = threadIdx.x & 31;
    const float* xr = x + (size_t)tok * DD;
    float a0 = 0.f, a1 = 0.f, a2 = 0.f, a3 = 0.f;
    for (int d = lane; d < DD; d += 32) {
        float xv  = xr[d];
        float4 wv = ((const float4*)rw)[d];       // router_w row d = 4 floats
        a0 += xv * wv.x; a1 += xv * wv.y; a2 += xv * wv.z; a3 += xv * wv.w;
    }
#pragma unroll
    for (int o = 16; o > 0; o >>= 1) {
        a0 += __shfl_down(a0, o);
        a1 += __shfl_down(a1, o);
        a2 += __shfl_down(a2, o);
        a3 += __shfl_down(a3, o);
    }
    if (lane == 0) r[tok] = make_float4(a0, a1, a2, a3);
}

// ---------------------------------------------------------------------------
// Kernel C: in-place inclusive prefix sum over S for each (b,e). 16 waves.
// ---------------------------------------------------------------------------
__global__ void k_scan(float* __restrict__ r) {
    int wid = threadIdx.x >> 5;          // 0..15 -> (b,e)
    int lane = threadIdx.x & 31;
    int b = wid >> 2, e = wid & 3;
    float* base = r + ((size_t)b * SS) * 4 + e;
    float carry = 0.f;
    for (int c = 0; c < SS / 32; ++c) {
        float v = base[(size_t)(c * 32 + lane) * 4];
#pragma unroll
        for (int o = 1; o < 32; o <<= 1) {
            float t = __shfl_up(v, o);
            if (lane >= o) v += t;
        }
        v += carry;
        base[(size_t)(c * 32 + lane) * 4] = v;
        carry = __shfl(v, 31);
    }
}

// ---------------------------------------------------------------------------
// Kernel D: softmax over E of prefix/(s+1); write top-1 gate + expert id.
// ---------------------------------------------------------------------------
__global__ void k_route(const float4* __restrict__ r, float* __restrict__ gate,
                        int* __restrict__ idx) {
    int tok = blockIdx.x * 256 + threadIdx.x;
    int s = tok % SS;
    float inv = 1.f / (float)(s + 1);
    float4 p = r[tok];
    float l[4] = {p.x * inv, p.y * inv, p.z * inv, p.w * inv};
    int im = 0; float m = l[0];
#pragma unroll
    for (int e = 1; e < 4; ++e) if (l[e] > m) { m = l[e]; im = e; }   // first-max tiebreak
    float sum = 0.f;
#pragma unroll
    for (int e = 0; e < 4; ++e) sum += __expf(l[e] - m);
    gate[tok] = 1.f / sum;               // exp(l[im]-m)/sum, l[im]==m
    idx[tok]  = im;
}

// ---------------------------------------------------------------------------
// Kernel E: ordered compaction into per-expert token lists (1 wave / expert).
// ---------------------------------------------------------------------------
__global__ void k_perm(const int* __restrict__ idx, int* __restrict__ perm,
                       int* __restrict__ counts) {
    int e = blockIdx.x;
    int lane = threadIdx.x;
    int* pl = perm + e * TT;
    int base = 0;
    for (int c = 0; c < TT / 32; ++c) {
        int t = c * 32 + lane;
        bool mine = (idx[t] == e);
        unsigned mask = (unsigned)__ballot(mine);
        if (mine) pl[base + __popc(mask & ((1u << lane) - 1u))] = t;
        base += __popc(mask);
    }
    int pad_end = (base + 15) & ~15;     // pad partial tile with token 0 (rows guarded)
    if (lane < pad_end - base) pl[base + lane] = 0;
    if (lane == 0) counts[e] = base;
}

// ---------------------------------------------------------------------------
// Kernel F: fused expert FFN. One block = (expert e, 16-token tile t).
// LDS: 32KB A-fragments (x tile, bf16) + 128KB h (16 x 4096 bf16) + row meta.
// Register-blocked N=4: one A fragment (2 ds_load_b128) feeds 4 WMMAs whose
// B fragments stream from L2 via 8 global_load_b128 (explicit AS1 pointers).
// Phase 1: h = gelu(x·w1 + b1)  (256 f-coltiles, 16 waves, 4 groups x K=32)
// Phase 2: y = (h·w2 + b2)*gate (64 d-coltiles, 16 waves, 1 group  x K=128)
// ---------------------------------------------------------------------------
__global__ __launch_bounds__(512, 1) void k_ffn(
    const float* __restrict__ x, const unsigned* __restrict__ w1sw,
    const unsigned* __restrict__ w2sw, const float* __restrict__ b1,
    const float* __restrict__ b2, const float* __restrict__ gate,
    const int* __restrict__ perm, const int* __restrict__ counts,
    float* __restrict__ out) {
    extern __shared__ char smem[];
    unsigned*       aFrag = (unsigned*)smem;                         // 32 KB
    unsigned short* hRow  = (unsigned short*)(smem + 32768);         // 128 KB
    int*            tokS  = (int*)(smem + 32768 + 131072);           // 16 ints
    float*          gateS = (float*)(smem + 32768 + 131072 + 64);    // 16 floats

    int e = blockIdx.x >> 9;
    int t = blockIdx.x & 511;
    int cnt = counts[e];
    if (t * 16 >= cnt) return;
    int nrow = cnt - t * 16; if (nrow > 16) nrow = 16;

    int tid = threadIdx.x;
    const int* pl = perm + e * TT + t * 16;
    if (tid < 16) { int tk = pl[tid]; tokS[tid] = tk; gateS[tid] = gate[tk]; }

    // ---- stage x tile as bf16 A-fragments: word i = [kt][lane][w] ----
    for (int i = tid; i < 8192; i += 512) {
        int w    = i & 7;
        int lane = (i >> 3) & 31;
        int kt   = i >> 8;
        int m    = lane & 15;
        int hi   = (lane >> 4) * 8;
        int k0   = kt * 32 + 2 * w + ((w >= 4) ? 8 : 0) + hi;
        int tk   = pl[m];
        f32x2 xv = *(gf2_t)(x + (size_t)tk * DD + k0);
        aFrag[i] = pack_bf2(xv.x, xv.y);
    }
    __syncthreads();

    int wid = tid >> 5;
    int lane = tid & 31;
    int nlo = lane & 15;
    int hi8 = (lane >> 4) * 8;

    // ---------------- phase 1: h = gelu(x @ w1 + b1) ----------------
    for (int g = 0; g < 4; ++g) {
        v8f acc[4] = {};
        gq_t bp[4];
#pragma unroll
        for (int j = 0; j < 4; ++j) {
            int ft = wid + 16 * (g * 4 + j);
            bp[j] = (gq_t)((const u32x4*)w1sw +
                ((size_t)(e * (FF / 16) + ft) * (DD / 32)) * 64 + lane * 2);
        }
        for (int kt = 0; kt < 32; ++kt) {
            Frag32 a;
            const u32x4* ap = (const u32x4*)(aFrag + (size_t)(kt * 32 + lane) * 8);
            a.q[0] = ap[0]; a.q[1] = ap[1];
#pragma unroll
            for (int j = 0; j < 4; ++j) {
                Frag32 b;
                b.q[0] = bp[j][0]; b.q[1] = bp[j][1];
                bp[j] += 64;
                acc[j] = __builtin_amdgcn_wmma_f32_16x16x32_bf16(
                    false, a.v, false, b.v, (short)0, acc[j], false, false);
            }
        }
#pragma unroll
        for (int j = 0; j < 4; ++j) {
            int ft = wid + 16 * (g * 4 + j);
            int f = ft * 16 + nlo;
            float bias = b1[e * FF + f];
#pragma unroll
            for (int rI = 0; rI < 8; ++rI) {
                float v = acc[j][rI] + bias;
                float u = 0.7978845608f * (v + 0.044715f * v * v * v);   // gelu (tanh)
                float gv = 0.5f * v * (1.0f + tanhf(u));
                int m = rI + hi8;
                hRow[(size_t)m * FF + f] = (unsigned short)bf_rne(gv);
            }
        }
    }
    __syncthreads();

    // ---------------- phase 2: y = (h @ w2 + b2) * gate -------------
    {
        v8f acc[4] = {};
        gq_t bp[4];
#pragma unroll
        for (int j = 0; j < 4; ++j) {
            int dt = wid + 16 * j;
            bp[j] = (gq_t)((const u32x4*)w2sw +
                ((size_t)(e * (DD / 16) + dt) * (FF / 32)) * 64 + lane * 2);
        }
        for (int kt = 0; kt < 128; ++kt) {
            Frag32 a;
            const unsigned short* hb = hRow + (size_t)nlo * FF + kt * 32 + hi8;
            a.q[0] = *(const u32x4*)(hb);
            a.q[1] = *(const u32x4*)(hb + 16);
#pragma unroll
            for (int j = 0; j < 4; ++j) {
                Frag32 b;
                b.q[0] = bp[j][0]; b.q[1] = bp[j][1];
                bp[j] += 64;
                acc[j] = __builtin_amdgcn_wmma_f32_16x16x32_bf16(
                    false, a.v, false, b.v, (short)0, acc[j], false, false);
            }
        }
#pragma unroll
        for (int j = 0; j < 4; ++j) {
            int dt = wid + 16 * j;
            int d = dt * 16 + nlo;
            float bias = b2[e * DD + d];
#pragma unroll
            for (int rI = 0; rI < 8; ++rI) {
                int m = rI + hi8;
                if (m < nrow) {
                    float y = (acc[j][rI] + bias) * gateS[m];
                    *(GAS float*)(out + (size_t)tokS[m] * DD + d) = y;
                }
            }
        }
    }
}

// ---------------------------------------------------------------------------
extern "C" void kernel_launch(void* const* d_in, const int* in_sizes, int n_in,
                              void* d_out, int out_size, void* d_ws, size_t ws_size,
                              hipStream_t stream) {
    const float* x   = (const float*)d_in[0];
    const float* rw  = (const float*)d_in[1];
    const float* w1  = (const float*)d_in[2];
    const float* b1  = (const float*)d_in[3];
    const float* w2  = (const float*)d_in[4];
    const float* b2  = (const float*)d_in[5];
    float* out = (float*)d_out;

    char* ws = (char*)d_ws;
    unsigned* w1sw  = (unsigned*)(ws);                  // 32 MB bf16 swizzled w1
    unsigned* w2sw  = (unsigned*)(ws + 33554432);       // 32 MB bf16 swizzled w2
    float*    rlog  = (float*)(ws + 67108864);          // 128 KB router logits/prefix
    float*    gate  = (float*)(ws + 67239936);          // 32 KB
    int*      idx   = (int*)(ws + 67272704);            // 32 KB
    int*      perm  = (int*)(ws + 67305472);            // 128 KB
    int*      cnts  = (int*)(ws + 67436544);            // 16 B

    // one-time per call: weights -> bf16 WMMA fragments (L2-resident thereafter)
    k_swizzle_w<<<4096, 256, 0, stream>>>(w1, w1sw, DD, FF);
    k_swizzle_w<<<4096, 256, 0, stream>>>(w2, w2sw, FF, DD);

    // routing
    k_router<<<TT / 8, 256, 0, stream>>>(x, rw, (float4*)rlog);
    k_scan<<<1, 512, 0, stream>>>(rlog);
    k_route<<<TT / 256, 256, 0, stream>>>((const float4*)rlog, gate, idx);
    k_perm<<<EE, 32, 0, stream>>>(idx, perm, cnts);

    // fused expert FFN (grid covers worst-case tiles; empty tiles early-exit)
    const int smem_bytes = 32768 + 131072 + 64 + 64;
    hipFuncSetAttribute((const void*)k_ffn,
                        hipFuncAttributeMaxDynamicSharedMemorySize, smem_bytes);
    k_ffn<<<EE * (TT / 16), 512, smem_bytes, stream>>>(
        x, w1sw, w2sw, b1, b2, gate, perm, cnts, out);
}
